// DecoderRNN_Attention_33861522162561
// MI455X (gfx1250) — compile-verified, matrix operand order checked by reading, not run on hardware
//
#include <hip/hip_runtime.h>
#include <hip/hip_bf16.h>

typedef __attribute__((ext_vector_type(16))) _Float16 v16h;
typedef __attribute__((ext_vector_type(8)))  _Float16 v8h;
typedef __attribute__((ext_vector_type(8)))  float    v8f;

#define Bn   128
#define Pn   196
#define ENCn 512
#define Hn   512
#define Vn_  32000
#define Ln   20
#define Tn   19

// ---------------------------------------------------------------------------
// f32 -> f16 2D convert with dest stride (for fused weight layouts)
__global__ void cvt2d_k(const float* __restrict__ src, _Float16* __restrict__ dst,
                        long long n, int cols, int dstride) {
    long long i = (long long)blockIdx.x * blockDim.x + threadIdx.x;
    if (i >= n) return;
    long long r = i / cols;
    int c = (int)(i % cols);
    dst[r * (long long)dstride + c] = (_Float16)src[i];
}

__global__ void copyf32_k(const float* __restrict__ s, float* __restrict__ d, int n) {
    int i = blockIdx.x * blockDim.x + threadIdx.x;
    if (i < n) d[i] = s[i];
}

__global__ void addvec_k(const float* __restrict__ a, const float* __restrict__ b,
                         float* __restrict__ d, int n) {
    int i = blockIdx.x * blockDim.x + threadIdx.x;
    if (i < n) d[i] = a[i] + b[i];
}

// fmean[b][j] = mean over P of features[b][p][j]
__global__ void fmean_k(const float* __restrict__ features, float* __restrict__ fmean) {
    int b = blockIdx.x;
    for (int j = threadIdx.x; j < ENCn; j += blockDim.x) {
        const float* f = features + ((size_t)b * Pn) * ENCn + j;
        float s = 0.f;
        for (int p = 0; p < Pn; ++p) s += f[(size_t)p * ENCn];
        fmean[b * ENCn + j] = s * (1.0f / (float)Pn);
    }
}

// naive fp32 GEMM (small, setup only): out[m][n] = sum_k A[m][k]*W[n][k] + bias[n]
__global__ void gemm32_naive_k(const float* __restrict__ A, const float* __restrict__ W,
                               const float* __restrict__ bias, float* __restrict__ out,
                               _Float16* __restrict__ out16, int N, int K) {
    int n = blockIdx.x * blockDim.x + threadIdx.x;
    int m = blockIdx.y;
    if (n >= N) return;
    const float* a = A + (size_t)m * K;
    const float* w = W + (size_t)n * K;
    float s = bias[n];
    for (int k = 0; k < K; ++k) s += a[k] * w[k];
    out[(size_t)m * N + n] = s;
    if (out16) out16[(size_t)m * N + n] = (_Float16)s;
}

// ---------------------------------------------------------------------------
// WMMA fragment helpers (ISA 16-bit A/B layouts, 32-wide K chunks)
struct Frags {
    v16h a;
    v16h b0, b1, b2, b3;
};

__device__ __forceinline__ v16h ld_frag(const _Float16* p) {
    v8h lo = *(const v8h*)(p);
    v8h hi = *(const v8h*)(p + 16);
    v16h r;
    #pragma unroll
    for (int i = 0; i < 8; ++i) { r[i] = lo[i]; r[8 + i] = hi[i]; }
    return r;
}

__device__ __forceinline__ void load_frags(const _Float16* Ap, const _Float16* Wp0,
                                           size_t ldw, int k0, Frags& f) {
    f.a  = ld_frag(Ap + k0);
    f.b0 = ld_frag(Wp0 + k0);
    f.b1 = ld_frag(Wp0 + 16 * ldw + k0);
    f.b2 = ld_frag(Wp0 + 32 * ldw + k0);
    f.b3 = ld_frag(Wp0 + 48 * ldw + k0);
}

__device__ __forceinline__ void mma_step(const Frags& f, v8f acc[4]) {
    acc[0] = __builtin_amdgcn_wmma_f32_16x16x32_f16(false, f.a, false, f.b0, (short)0, acc[0], false, false);
    acc[1] = __builtin_amdgcn_wmma_f32_16x16x32_f16(false, f.a, false, f.b1, (short)0, acc[1], false, false);
    acc[2] = __builtin_amdgcn_wmma_f32_16x16x32_f16(false, f.a, false, f.b2, (short)0, acc[2], false, false);
    acc[3] = __builtin_amdgcn_wmma_f32_16x16x32_f16(false, f.a, false, f.b3, (short)0, acc[3], false, false);
}

// ---------------------------------------------------------------------------
// WMMA f16 GEMM: C[M,N] = A[M,K] * W[N,K]^T + bias.
// Each wave32 computes a 16x64 tile; software-pipelined 2-stage K loop so
// next-chunk global_load_b128s overlap the current chunk's 4 v_wmma.
// MODE 0: write C fp32 row-major.
// MODE 1: masked prediction write to d_out at ((row*T + t)*N + col), 0 if inactive.
template <int MODE>
__global__ __launch_bounds__(256)
void gemm_wmma_k(const _Float16* __restrict__ A, int lda,
                 const _Float16* __restrict__ W, int ldw,
                 const float* __restrict__ bias,
                 float* __restrict__ out,
                 int M, int N, int K,
                 const int* __restrict__ lengths, int t) {
    const int wave = (blockIdx.x * blockDim.x + threadIdx.x) >> 5;
    const int lane = threadIdx.x & 31;
    const int nTiles = N >> 6;                 // 64 cols per wave
    const int mTile  = wave / nTiles;
    const int nTile  = wave % nTiles;
    if (mTile * 16 >= M) return;
    const int mBase = mTile * 16;
    const int nBase = nTile * 64;

    const int row = mBase + (lane & 15);
    const int kb  = (lane < 16) ? 0 : 8;
    const _Float16* Ap  = A + (size_t)row * lda + kb;
    const _Float16* Wp0 = W + (size_t)(nBase + (lane & 15)) * ldw + kb;

    v8f acc[4];
    #pragma unroll
    for (int j = 0; j < 4; ++j) acc[j] = (v8f){0.f,0.f,0.f,0.f,0.f,0.f,0.f,0.f};

    // K/32 chunks; all shapes here have an even chunk count (16 or 48).
    Frags fA, fB;
    load_frags(Ap, Wp0, (size_t)ldw, 0, fA);
    int k0 = 32;
    const int pairs = (K / 32 - 2) / 2;
    for (int it = 0; it < pairs; ++it) {
        load_frags(Ap, Wp0, (size_t)ldw, k0, fB);
        mma_step(fA, acc);
        load_frags(Ap, Wp0, (size_t)ldw, k0 + 32, fA);
        mma_step(fB, acc);
        k0 += 64;
    }
    load_frags(Ap, Wp0, (size_t)ldw, k0, fB);
    mma_step(fA, acc);
    mma_step(fB, acc);

    // C/D layout: VGPR r -> row mBase + r (+8 for lanes 16..31), col = nBase + lane&15.
    const int rbase = mBase + ((lane < 16) ? 0 : 8);
    const int col0  = nBase + (lane & 15);
    #pragma unroll
    for (int j = 0; j < 4; ++j) {
        const int col = col0 + j * 16;
        const float bv = bias[col];
        #pragma unroll
        for (int r = 0; r < 8; ++r) {
            const int rr = rbase + r;
            float v = acc[j][r] + bv;
            if (MODE == 0) {
                out[(size_t)rr * N + col] = v;
            } else {
                const bool act = t < (lengths[rr] - 1);
                out[((size_t)rr * Tn + t) * (size_t)N + col] = act ? v : 0.f;
            }
        }
    }
}

// ---------------------------------------------------------------------------
// Fused attention per batch row: e -> softmax -> alpha (masked to d_out),
// awe = alpha @ features, gated; assembles z16 = [emb | gate*awe | h] (f16).
__global__ __launch_bounds__(256)
void attn_k(const float* __restrict__ att1, const float* __restrict__ att2g,
            const float* __restrict__ features, const float* __restrict__ w_full,
            const float* __restrict__ b_full, const float* __restrict__ h,
            const float* __restrict__ embed_w, const int* __restrict__ captions,
            const int* __restrict__ lengths,
            _Float16* __restrict__ z16, float* __restrict__ alphas_out, int t) {
    __shared__ float s_att2[ENCn];
    __shared__ float s_e[Pn];
    __shared__ float s_red[8];
    const int b = blockIdx.x;
    const int tid = threadIdx.x;
    const int lane = tid & 31, warp = tid >> 5;

    for (int j = tid; j < ENCn; j += 256) s_att2[j] = att2g[b * 1024 + j];
    __syncthreads();

    // e[p] = relu(att1[b,p,:] + att2) . w_full + b_full
    for (int p = warp; p < Pn; p += 8) {
        const float* arow = att1 + ((size_t)b * Pn + p) * ENCn;
        float s = 0.f;
        for (int a = lane; a < ENCn; a += 32) {
            float v = arow[a] + s_att2[a];
            v = v > 0.f ? v : 0.f;
            s += v * w_full[a];
        }
        #pragma unroll
        for (int o = 16; o > 0; o >>= 1) s += __shfl_xor(s, o, 32);
        if (lane == 0) s_e[p] = s + b_full[0];
    }
    __syncthreads();

    // softmax over P
    float m = -3.4e38f;
    for (int p = tid; p < Pn; p += 256) m = fmaxf(m, s_e[p]);
    #pragma unroll
    for (int o = 16; o > 0; o >>= 1) m = fmaxf(m, __shfl_xor(m, o, 32));
    if (lane == 0) s_red[warp] = m;
    __syncthreads();
    if (tid == 0) {
        float mm = s_red[0];
        for (int i = 1; i < 8; ++i) mm = fmaxf(mm, s_red[i]);
        s_red[0] = mm;
    }
    __syncthreads();
    m = s_red[0];
    __syncthreads();

    float sum = 0.f;
    for (int p = tid; p < Pn; p += 256) {
        float ev = __expf(s_e[p] - m);
        s_e[p] = ev;
        sum += ev;
    }
    #pragma unroll
    for (int o = 16; o > 0; o >>= 1) sum += __shfl_xor(sum, o, 32);
    if (lane == 0) s_red[warp] = sum;
    __syncthreads();
    if (tid == 0) {
        float ss = 0.f;
        for (int i = 0; i < 8; ++i) ss += s_red[i];
        s_red[0] = ss;
    }
    __syncthreads();
    const float inv = 1.0f / s_red[0];
    const bool act = t < (lengths[b] - 1);
    for (int p = tid; p < Pn; p += 256) {
        float a = s_e[p] * inv;
        s_e[p] = a;
        alphas_out[((size_t)b * Tn + t) * Pn + p] = act ? a : 0.f;
    }
    __syncthreads();

    // awe + gate + z assembly
    const int cap = captions[b * Ln + t];
    _Float16* z = z16 + (size_t)b * 1536;
    for (int j = tid; j < ENCn; j += 256) {
        const float* f = features + ((size_t)b * Pn) * ENCn + j;
        float aw = 0.f;
        for (int p = 0; p < Pn; ++p) aw += s_e[p] * f[(size_t)p * ENCn];
        const float gp = att2g[b * 1024 + 512 + j];
        const float g = 1.0f / (1.0f + __expf(-gp));
        z[j]        = (_Float16)embed_w[(size_t)cap * ENCn + j];
        z[512 + j]  = (_Float16)(aw * g);
        z[1024 + j] = (_Float16)h[b * Hn + j];
    }
}

// ---------------------------------------------------------------------------
// Elementwise LSTM cell update (in place), emits h2 as f16 for next GEMMs.
__global__ void lstm_k(const float* __restrict__ gates, float* __restrict__ h,
                       float* __restrict__ c, _Float16* __restrict__ h16,
                       const int* __restrict__ lengths, int t) {
    int i = blockIdx.x * blockDim.x + threadIdx.x;
    if (i >= Bn * Hn) return;
    const int b = i >> 9;
    const int j = i & 511;
    const float* g = gates + (size_t)b * 2048;
    const float sig_i = 1.f / (1.f + __expf(-g[j]));
    const float sig_f = 1.f / (1.f + __expf(-g[512 + j]));
    const float tg    = tanhf(g[1024 + j]);
    const float sig_o = 1.f / (1.f + __expf(-g[1536 + j]));
    const float cn = sig_f * c[i] + sig_i * tg;
    const float hn = sig_o * tanhf(cn);
    const bool act = t < (lengths[b] - 1);
    const float h2 = act ? hn : h[i];
    const float c2 = act ? cn : c[i];
    h[i] = h2;
    c[i] = c2;
    h16[i] = (_Float16)h2;
}

// ---------------------------------------------------------------------------
extern "C" void kernel_launch(void* const* d_in, const int* in_sizes, int n_in,
                              void* d_out, int out_size, void* d_ws, size_t ws_size,
                              hipStream_t stream) {
    (void)in_sizes; (void)n_in; (void)out_size; (void)ws_size;
    const float* features  = (const float*)d_in[0];
    const int*   captions  = (const int*)d_in[1];
    const int*   lengths   = (const int*)d_in[2];
    const float* w_enc_att = (const float*)d_in[3];
    const float* b_enc_att = (const float*)d_in[4];
    const float* w_dec_att = (const float*)d_in[5];
    const float* b_dec_att = (const float*)d_in[6];
    const float* w_full    = (const float*)d_in[7];
    const float* b_full    = (const float*)d_in[8];
    const float* embed_w   = (const float*)d_in[9];
    const float* w_ih      = (const float*)d_in[10];
    const float* b_ih      = (const float*)d_in[11];
    const float* w_hh      = (const float*)d_in[12];
    const float* b_hh      = (const float*)d_in[13];
    const float* w_init_h  = (const float*)d_in[14];
    const float* b_init_h  = (const float*)d_in[15];
    const float* w_init_c  = (const float*)d_in[16];
    const float* b_init_c  = (const float*)d_in[17];
    const float* w_beta    = (const float*)d_in[18];
    const float* b_beta    = (const float*)d_in[19];
    const float* w_fc      = (const float*)d_in[20];
    const float* b_fc      = (const float*)d_in[21];

    float* preds_out  = (float*)d_out;
    float* alphas_out = preds_out + (size_t)Bn * Tn * Vn_;

    // workspace carving (256B aligned)
    char* ws = (char*)d_ws;
    size_t off = 0;
    auto carve = [&](size_t bytes) -> void* {
        off = (off + 255) & ~(size_t)255;
        void* p = ws + off;
        off += bytes;
        return p;
    };
    const size_t BP = (size_t)Bn * Pn;                       // 25088
    _Float16* feat16  = (_Float16*)carve(BP * ENCn * 2);
    _Float16* wenc16  = (_Float16*)carve((size_t)512 * 512 * 2);
    _Float16* wdab16  = (_Float16*)carve((size_t)1024 * 512 * 2);
    _Float16* wcomb16 = (_Float16*)carve((size_t)2048 * 1536 * 2);
    _Float16* wfc16   = (_Float16*)carve((size_t)Vn_ * 512 * 2);
    float*    att1    = (float*)carve(BP * ENCn * 4);
    float*    fmean   = (float*)carve((size_t)Bn * ENCn * 4);
    float*    hbuf    = (float*)carve((size_t)Bn * Hn * 4);
    float*    cbuf    = (float*)carve((size_t)Bn * Hn * 4);
    _Float16* h16     = (_Float16*)carve((size_t)Bn * Hn * 2);
    float*    att2g   = (float*)carve((size_t)Bn * 1024 * 4);
    float*    gates   = (float*)carve((size_t)Bn * 2048 * 4);
    _Float16* z16     = (_Float16*)carve((size_t)Bn * 1536 * 2);
    float*    bdab    = (float*)carve(1024 * 4);
    float*    bcomb   = (float*)carve(2048 * 4);

    auto cgrid = [](long long n) { return (unsigned)((n + 255) / 256); };
    auto gemm_blocks = [](int M, int N) {
        long long tiles = (long long)((M + 15) / 16) * (N / 64);
        return (unsigned)((tiles + 7) / 8);
    };

    // ---- one-time conversions / fused weight layouts ----
    cvt2d_k<<<cgrid(BP * ENCn), 256, 0, stream>>>(features, feat16, BP * ENCn, 512, 512);
    cvt2d_k<<<cgrid(512LL * 512), 256, 0, stream>>>(w_enc_att, wenc16, 512LL * 512, 512, 512);
    cvt2d_k<<<cgrid(512LL * 512), 256, 0, stream>>>(w_dec_att, wdab16, 512LL * 512, 512, 512);
    cvt2d_k<<<cgrid(512LL * 512), 256, 0, stream>>>(w_beta, wdab16 + 512 * 512, 512LL * 512, 512, 512);
    cvt2d_k<<<cgrid(2048LL * 1024), 256, 0, stream>>>(w_ih, wcomb16, 2048LL * 1024, 1024, 1536);
    cvt2d_k<<<cgrid(2048LL * 512), 256, 0, stream>>>(w_hh, wcomb16 + 1024, 2048LL * 512, 512, 1536);
    cvt2d_k<<<cgrid((long long)Vn_ * 512), 256, 0, stream>>>(w_fc, wfc16, (long long)Vn_ * 512, 512, 512);
    addvec_k<<<8, 256, 0, stream>>>(b_ih, b_hh, bcomb, 2048);
    copyf32_k<<<2, 256, 0, stream>>>(b_dec_att, bdab, 512);
    copyf32_k<<<2, 256, 0, stream>>>(b_beta, bdab + 512, 512);

    // ---- initial state ----
    fmean_k<<<Bn, 256, 0, stream>>>(features, fmean);
    gemm32_naive_k<<<dim3(2, Bn), 256, 0, stream>>>(fmean, w_init_h, b_init_h, hbuf, h16, 512, 512);
    gemm32_naive_k<<<dim3(2, Bn), 256, 0, stream>>>(fmean, w_init_c, b_init_c, cbuf, nullptr, 512, 512);

    // ---- att1 = features @ w_enc_att.T + b (25088 x 512, K=512), WMMA ----
    gemm_wmma_k<0><<<gemm_blocks((int)BP, 512), 256, 0, stream>>>(
        feat16, 512, wenc16, 512, b_enc_att, att1, (int)BP, 512, 512, nullptr, 0);

    // ---- 19 sequential decode steps ----
    for (int t = 0; t < Tn; ++t) {
        // att2 | gate-preact : h @ [w_dec_att; w_beta].T  (128 x 1024, K=512)
        gemm_wmma_k<0><<<gemm_blocks(Bn, 1024), 256, 0, stream>>>(
            h16, 512, wdab16, 512, bdab, att2g, Bn, 1024, 512, nullptr, 0);

        attn_k<<<Bn, 256, 0, stream>>>(att1, att2g, features, w_full, b_full, hbuf,
                                       embed_w, captions, lengths, z16, alphas_out, t);

        // gates = [emb|awe|h] @ [w_ih|w_hh].T + (b_ih+b_hh)  (128 x 2048, K=1536)
        gemm_wmma_k<0><<<gemm_blocks(Bn, 2048), 256, 0, stream>>>(
            z16, 1536, wcomb16, 1536, bcomb, gates, Bn, 2048, 1536, nullptr, 0);

        lstm_k<<<cgrid((long long)Bn * Hn), 256, 0, stream>>>(gates, hbuf, cbuf, h16, lengths, t);

        // predictions = h_new @ w_fc.T + b_fc, masked, straight to d_out
        gemm_wmma_k<1><<<gemm_blocks(Bn, Vn_), 256, 0, stream>>>(
            h16, 512, wfc16, 512, b_fc, preds_out, Bn, Vn_, 512, lengths, t);
    }
}
